// queue_ila_74947179316050
// MI455X (gfx1250) — matrix operand order repaired and checked by reading, not compiled, not run on hardware
//
#include <hip/hip_runtime.h>
#include <hip/hip_bf16.h>
#include <stdint.h>

#define DIM 256
#define BATCH 256
#define KQ 65536
#define NUM_CLASSES 126
#define TOPN 32
#define RANKK 4
#define TAU 0.007f
#define INV_TAU (1.0f / TAU)

#define N_CHUNK 4096
#define CHUNKS (KQ / N_CHUNK)            // 16
#define ROWGROUPS (BATCH / 16)           // 16
#define TILES_PER_CHUNK (N_CHUNK / 16)   // 256
#define WAVES 8
#define TILES_PER_WAVE (TILES_PER_CHUNK / WAVES) // 32

typedef __attribute__((ext_vector_type(16))) __bf16 v16bf;
typedef __attribute__((ext_vector_type(8)))  float  v8f;
typedef __attribute__((__vector_size__(16))) int    v4i;

union BFrag { v16bf v; uint4 q[2]; unsigned short u[16]; };

// CDNA5 async global->LDS path (ASYNCcnt-tracked), guarded for toolchain support.
#if defined(__has_builtin)
#if __has_builtin(__builtin_amdgcn_global_load_async_to_lds_b128) && \
    __has_builtin(__builtin_amdgcn_s_wait_asynccnt)
#define USE_ASYNC_LDS 1
#endif
#endif

#define AS1 __attribute__((address_space(1)))
#define AS3 __attribute__((address_space(3)))

// ---------------- workspace layout (bytes) ----------------
constexpr size_t OFF_QN2  = 0;                                    // KQ f32
constexpr size_t OFF_QLAB = OFF_QN2  + (size_t)KQ * 4;            // KQ i32
constexpr size_t OFF_TN2  = OFF_QLAB + (size_t)KQ * 4;            // BATCH f32
constexpr size_t OFF_RMAX = OFF_TN2  + (size_t)BATCH * 4;         // BATCH*CHUNKS f32
constexpr size_t OFF_TOPK = OFF_RMAX + (size_t)BATCH * CHUNKS * 4;        // BATCH*CHUNKS*TOPN u64
constexpr size_t OFF_PSE  = OFF_TOPK + (size_t)BATCH * CHUNKS * TOPN * 8; // BATCH i32
constexpr size_t OFF_MROW = OFF_PSE  + (size_t)BATCH * 4;         // BATCH f32
constexpr size_t OFF_SALL = OFF_MROW + (size_t)BATCH * 4;         // BATCH*CHUNKS f32
constexpr size_t OFF_SPOS = OFF_SALL + (size_t)BATCH * CHUNKS * 4;// BATCH*CHUNKS f32

// ---------------- helpers ----------------
__device__ __forceinline__ const float* q_eff_row(const float* queue, const float* feats,
                                                  int j, int qptr) {
  unsigned u = (unsigned)(j - qptr);
  return (u < (unsigned)BATCH) ? (feats + (size_t)u * DIM) : (queue + (size_t)j * DIM);
}
__device__ __forceinline__ float bf_hi_f(float f) {
  return __uint_as_float(__float_as_uint(f) & 0xFFFF0000u);
}

// Compute one 16x16 tile of similarities via bf16-split WMMA.
// Returns v8f: element e is sim for row M=(lane>>4)*8+e, col N=col0+(lane&15).
__device__ __forceinline__ v8f tile_sims(const unsigned short sAhi[16][DIM],
                                         const unsigned short sAlo[16][DIM],
                                         const float* queue, const float* feats, int qptr,
                                         const float* qn2, const float* sTn2,
                                         int col0, int lane) {
  int colg = col0 + (lane & 15);
  const float* qp = q_eff_row(queue, feats, colg, qptr);
  // Prefetch this wave's next B tile rows (stride = WAVES*16 columns).
  {
    int pcol = colg + WAVES * 16;
    if (pcol < KQ) __builtin_prefetch(q_eff_row(queue, feats, pcol, qptr), 0, 0);
  }
  int kh   = lane >> 4;
  int arow = lane & 15;

  v8f c = {0.f, 0.f, 0.f, 0.f, 0.f, 0.f, 0.f, 0.f};
  #pragma unroll
  for (int kt = 0; kt < DIM / 32; ++kt) {
    int kbase = kt * 32;
    // A fragments: lane<16 holds K {0..15}, lane>=16 holds K {16..31}
    int aklo = kbase + kh * 16;
    int akhi = kbase + kh * 16 + 8;
    BFrag ah, al;
    ah.q[0] = *(const uint4*)&sAhi[arow][aklo];
    ah.q[1] = *(const uint4*)&sAhi[arow][akhi];
    al.q[0] = *(const uint4*)&sAlo[arow][aklo];
    al.q[1] = *(const uint4*)&sAlo[arow][akhi];
    // B fragment: lane<16 holds N=lane, K=kbase..kbase+15; lane>=16: K=kbase+16..+31
    int bk = kbase + kh * 16;
    const float4* qp4 = (const float4*)(qp + bk);
    BFrag bh, bl;
    #pragma unroll
    for (int q4 = 0; q4 < 4; ++q4) {
      float4 f = qp4[q4];
      float vv[4] = {f.x, f.y, f.z, f.w};
      #pragma unroll
      for (int e = 0; e < 4; ++e) {
        float x  = vv[e];
        float xh = bf_hi_f(x);
        bh.u[q4 * 4 + e] = (unsigned short)(__float_as_uint(x) >> 16);
        bl.u[q4 * 4 + e] = (unsigned short)(__float_as_uint(x - xh) >> 16);
      }
    }
    c = __builtin_amdgcn_wmma_f32_16x16x32_bf16(false, al.v, false, bh.v, (short)0, c, false, false);
    c = __builtin_amdgcn_wmma_f32_16x16x32_bf16(false, ah.v, false, bl.v, (short)0, c, false, false);
    c = __builtin_amdgcn_wmma_f32_16x16x32_bf16(false, ah.v, false, bh.v, (short)0, c, false, false);
  }
  float qcol = qn2[colg];
  v8f sims;
  #pragma unroll
  for (int e = 0; e < 8; ++e) {
    int   r  = kh * 8 + e;
    float d2 = fmaxf(sTn2[r] + qcol - 2.f * c[e], 0.f);
    sims[e]  = 1.f / (1.f + sqrtf(d2));
  }
  return sims;
}

#ifdef USE_ASYNC_LDS
// Stage the 16x256-f32 A tile into LDS with GLOBAL_LOAD_ASYNC_TO_LDS_B128,
// wait on ASYNCcnt, then split f32 -> bf16 hi/lo from LDS.
__device__ __forceinline__ void load_A_tile_async(unsigned short sAhi[16][DIM],
                                                  unsigned short sAlo[16][DIM],
                                                  float sF[16][DIM], float sTn2[16],
                                                  const float* feats, const float* tn2,
                                                  int row0) {
  const float* g = feats + (size_t)(BATCH + row0) * DIM;  // contiguous 16KB
  #pragma unroll
  for (int i = 0; i < (16 * DIM) / (256 * 4); ++i) {       // 4 x b128 per thread
    int idx = (int)threadIdx.x + i * 256;                  // 16B granules
    __builtin_amdgcn_global_load_async_to_lds_b128(
        (AS1 v4i*)(uintptr_t)(g + idx * 4),
        (AS3 v4i*)(void*)(&sF[0][0] + idx * 4), 0, 0);
  }
  __builtin_amdgcn_s_wait_asynccnt(0);
  __syncthreads();
  for (int idx = threadIdx.x; idx < 16 * DIM; idx += blockDim.x) {
    int r = idx >> 8, k = idx & 255;
    float x  = sF[r][k];
    float xh = bf_hi_f(x);
    sAhi[r][k] = (unsigned short)(__float_as_uint(x) >> 16);
    sAlo[r][k] = (unsigned short)(__float_as_uint(x - xh) >> 16);
  }
  if (threadIdx.x < 16) sTn2[threadIdx.x] = tn2[row0 + threadIdx.x];
}
#else
__device__ __forceinline__ void load_A_tile(unsigned short sAhi[16][DIM],
                                            unsigned short sAlo[16][DIM],
                                            float sTn2[16], const float* feats,
                                            const float* tn2, int row0) {
  for (int idx = threadIdx.x; idx < 16 * DIM; idx += blockDim.x) {
    int r = idx >> 8, k = idx & 255;
    float x  = feats[(size_t)(BATCH + row0 + r) * DIM + k];
    float xh = bf_hi_f(x);
    sAhi[r][k] = (unsigned short)(__float_as_uint(x) >> 16);
    sAlo[r][k] = (unsigned short)(__float_as_uint(x - xh) >> 16);
  }
  if (threadIdx.x < 16) sTn2[threadIdx.x] = tn2[row0 + threadIdx.x];
}
#endif

// ---------------- kernel 0: norms + effective labels ----------------
__global__ __launch_bounds__(256) void prep_kernel(const float* feats, const int* slab,
                                                   const float* queue, const int* qlabin,
                                                   const int* qptrp, float* qn2, int* qlab,
                                                   float* tn2) {
  int j = blockIdx.x * blockDim.x + threadIdx.x;
  int qptr = *qptrp;
  if (j < KQ) {
    const float4* r4 = (const float4*)q_eff_row(queue, feats, j, qptr);
    float s = 0.f;
    for (int k = 0; k < DIM / 4; ++k) {
      float4 v = r4[k];
      s += v.x * v.x + v.y * v.y + v.z * v.z + v.w * v.w;
    }
    qn2[j] = s;
    unsigned u = (unsigned)(j - qptr);
    qlab[j] = (u < (unsigned)BATCH) ? slab[u] : qlabin[j];
  }
  if (j < BATCH) {
    const float4* r4 = (const float4*)(feats + (size_t)(BATCH + j) * DIM);
    float s = 0.f;
    for (int k = 0; k < DIM / 4; ++k) {
      float4 v = r4[k];
      s += v.x * v.x + v.y * v.y + v.z * v.z + v.w * v.w;
    }
    tn2[j] = s;
  }
}

// ---------------- pass A: GEMM + per-chunk top-32 + row max ----------------
__global__ __launch_bounds__(256) void passA_kernel(const float* feats, const float* queue,
                                                    const int* qptrp, const float* qn2,
                                                    const int* qlab, const float* tn2,
                                                    float* rowmax, unsigned long long* topk) {
  __shared__ __align__(16) unsigned short sAhi[16][DIM];
  __shared__ __align__(16) unsigned short sAlo[16][DIM];
  __shared__ float sTn2[16];
  __shared__ float sSim[WAVES][16][16];
  __shared__ int   sLab[WAVES][16];
  __shared__ unsigned long long wtop[WAVES][16][TOPN];
#ifdef USE_ASYNC_LDS
  __shared__ __align__(16) float sF[16][DIM];
#endif

  int tid = threadIdx.x, lane = tid & 31, wave = tid >> 5;
  int chunk = blockIdx.x, row0 = blockIdx.y * 16;
  int colchunk = chunk * N_CHUNK;
  int qptr = *qptrp;

  for (int i = tid; i < WAVES * 16 * TOPN; i += blockDim.x)
    ((unsigned long long*)wtop)[i] = 0ull;
#ifdef USE_ASYNC_LDS
  load_A_tile_async(sAhi, sAlo, sF, sTn2, feats, tn2, row0);
#else
  load_A_tile(sAhi, sAlo, sTn2, feats, tn2, row0);
#endif
  __syncthreads();

  for (int t = 0; t < TILES_PER_WAVE; ++t) {
    int tile = t * WAVES + wave;
    int col0 = colchunk + tile * 16;
    v8f sims = tile_sims(sAhi, sAlo, queue, feats, qptr, qn2, sTn2, col0, lane);
    int kh = lane >> 4;
    #pragma unroll
    for (int e = 0; e < 8; ++e) sSim[wave][kh * 8 + e][lane & 15] = sims[e];
    if (lane < 16) sLab[wave][lane] = qlab[col0 + lane];
    __syncthreads();
    if (lane < 16) {
      unsigned long long* lst = wtop[wave][lane];
      #pragma unroll 1
      for (int cc = 0; cc < 16; ++cc) {
        float s = sSim[wave][lane][cc];
        unsigned long long key =
            ((unsigned long long)__float_as_uint(s) << 32) | (unsigned)sLab[wave][cc];
        if (key > lst[TOPN - 1]) {
          int p = TOPN - 1;
          while (p > 0 && lst[p - 1] < key) { lst[p] = lst[p - 1]; --p; }
          lst[p] = key;
        }
      }
    }
    __syncthreads();
  }

  if (tid < 16) {  // 8-way merge of wave lists per row
    int r = tid;
    int idx[WAVES];
    #pragma unroll
    for (int w = 0; w < WAVES; ++w) idx[w] = 0;
    size_t base = ((size_t)(row0 + r) * CHUNKS + chunk) * TOPN;
    float mx = 0.f;
    for (int o = 0; o < TOPN; ++o) {
      int bw = 0; unsigned long long bk = 0ull;
      #pragma unroll
      for (int w = 0; w < WAVES; ++w) {
        unsigned long long k = (idx[w] < TOPN) ? wtop[w][r][idx[w]] : 0ull;
        if (k > bk) { bk = k; bw = w; }
      }
      idx[bw]++;
      topk[base + o] = bk;
      if (o == 0) mx = __uint_as_float((unsigned)(bk >> 32));
    }
    rowmax[(size_t)(row0 + r) * CHUNKS + chunk] = mx;
  }
}

// ---------------- pass B: merge chunks, pseudo label + global max ----------------
__global__ __launch_bounds__(256) void passB_kernel(const unsigned long long* topk,
                                                    const float* rowmax, int* pseudo,
                                                    float* Mrow) {
  __shared__ unsigned char cnt[BATCH][128];
  int i = threadIdx.x;
  for (int c = 0; c < 128; ++c) cnt[i][c] = 0;  // private row, no barrier needed

  unsigned long long head[CHUNKS]; int idx[CHUNKS];
  const unsigned long long* base = topk + (size_t)i * CHUNKS * TOPN;
  #pragma unroll
  for (int cc = 0; cc < CHUNKS; ++cc) { idx[cc] = 0; head[cc] = base[(size_t)cc * TOPN]; }

  int firstK = -1;
  float mx = 0.f;
  for (int o = 0; o < TOPN; ++o) {
    int b = 0; unsigned long long bk = 0ull;
    #pragma unroll
    for (int cc = 0; cc < CHUNKS; ++cc)
      if (head[cc] > bk) { bk = head[cc]; b = cc; }
    idx[b]++;
    head[b] = (idx[b] < TOPN) ? base[(size_t)b * TOPN + idx[b]] : 0ull;
    if (o == 0) mx = __uint_as_float((unsigned)(bk >> 32));
    int lab = (int)(unsigned)(bk & 0xFFFFFFFFu);
    unsigned char c = ++cnt[i][lab];
    if (c == RANKK && firstK < 0) firstK = lab;
  }
  int ps;
  if (firstK >= 0) ps = firstK;
  else {
    int best = 0, bc = -1;
    for (int c = 0; c < NUM_CLASSES; ++c) {
      int v = cnt[i][c];
      if (v > bc) { bc = v; best = c; }
    }
    ps = best;
  }
  pseudo[i] = ps;
  Mrow[i] = mx;
  (void)rowmax;
}

// ---------------- pass C: GEMM + shifted exp sums (all / positives) ----------------
__global__ __launch_bounds__(256) void passC_kernel(const float* feats, const float* queue,
                                                    const int* qptrp, const float* qn2,
                                                    const int* qlab, const float* tn2,
                                                    const int* pseudo, const float* Mrow,
                                                    float* sumall, float* sumpos) {
  __shared__ __align__(16) unsigned short sAhi[16][DIM];
  __shared__ __align__(16) unsigned short sAlo[16][DIM];
  __shared__ float sTn2[16];
  __shared__ float sM[16];
  __shared__ int   sPs[16];
  __shared__ float bAll[16], bPos[16];
#ifdef USE_ASYNC_LDS
  __shared__ __align__(16) float sF[16][DIM];
#endif

  int tid = threadIdx.x, lane = tid & 31, wave = tid >> 5;
  int chunk = blockIdx.x, row0 = blockIdx.y * 16;
  int colchunk = chunk * N_CHUNK;
  int qptr = *qptrp;

#ifdef USE_ASYNC_LDS
  load_A_tile_async(sAhi, sAlo, sF, sTn2, feats, tn2, row0);
#else
  load_A_tile(sAhi, sAlo, sTn2, feats, tn2, row0);
#endif
  if (tid < 16) {
    sM[tid]   = Mrow[row0 + tid];
    sPs[tid]  = pseudo[row0 + tid];
    bAll[tid] = 0.f;
    bPos[tid] = 0.f;
  }
  __syncthreads();

  float accA[8] = {0.f, 0.f, 0.f, 0.f, 0.f, 0.f, 0.f, 0.f};
  float accP[8] = {0.f, 0.f, 0.f, 0.f, 0.f, 0.f, 0.f, 0.f};
  int kh = lane >> 4;

  for (int t = 0; t < TILES_PER_WAVE; ++t) {
    int tile = t * WAVES + wave;
    int col0 = colchunk + tile * 16;
    v8f sims = tile_sims(sAhi, sAlo, queue, feats, qptr, qn2, sTn2, col0, lane);
    int labc = qlab[col0 + (lane & 15)];
    #pragma unroll
    for (int e = 0; e < 8; ++e) {
      int r = kh * 8 + e;
      float ex = __expf((sims[e] - sM[r]) * INV_TAU);
      accA[e] += ex;
      if (labc == sPs[r]) accP[e] += ex;
    }
  }
  #pragma unroll
  for (int e = 0; e < 8; ++e) {
    int r = kh * 8 + e;
    atomicAdd(&bAll[r], accA[e]);
    atomicAdd(&bPos[r], accP[e]);
  }
  __syncthreads();
  if (tid < 16) {
    sumall[(size_t)(row0 + tid) * CHUNKS + chunk] = bAll[tid];
    sumpos[(size_t)(row0 + tid) * CHUNKS + chunk] = bPos[tid];
  }
}

// ---------------- pass D: final loss ----------------
__global__ __launch_bounds__(256) void passD_kernel(const float* sumall, const float* sumpos,
                                                    float* out) {
  __shared__ float red[BATCH];
  int i = threadIdx.x;
  float sa = 0.f, sp = 0.f;
  for (int c = 0; c < CHUNKS; ++c) {
    sa += sumall[(size_t)i * CHUNKS + c];
    sp += sumpos[(size_t)i * CHUNKS + c];
  }
  red[i] = logf(sa) - logf(sp);  // shifts cancel
  __syncthreads();
  for (int s = BATCH / 2; s > 0; s >>= 1) {
    if (i < s) red[i] += red[i + s];
    __syncthreads();
  }
  if (i == 0) out[0] = red[0] / (float)BATCH;
}

extern "C" void kernel_launch(void* const* d_in, const int* in_sizes, int n_in,
                              void* d_out, int out_size, void* d_ws, size_t ws_size,
                              hipStream_t stream) {
  (void)in_sizes; (void)n_in; (void)out_size; (void)ws_size;
  const float* feats  = (const float*)d_in[0];
  const int*   slab   = (const int*)d_in[1];
  const float* queue  = (const float*)d_in[3];   // d_in[2] target_labels unused
  const int*   qlabin = (const int*)d_in[4];
  const int*   qptr   = (const int*)d_in[5];

  char* ws = (char*)d_ws;
  float*              qn2    = (float*)(ws + OFF_QN2);
  int*                qlab   = (int*)(ws + OFF_QLAB);
  float*              tn2    = (float*)(ws + OFF_TN2);
  float*              rowmax = (float*)(ws + OFF_RMAX);
  unsigned long long* topk   = (unsigned long long*)(ws + OFF_TOPK);
  int*                pseudo = (int*)(ws + OFF_PSE);
  float*              Mrow   = (float*)(ws + OFF_MROW);
  float*              sumall = (float*)(ws + OFF_SALL);
  float*              sumpos = (float*)(ws + OFF_SPOS);

  prep_kernel<<<KQ / 256, 256, 0, stream>>>(feats, slab, queue, qlabin, qptr, qn2, qlab, tn2);
  dim3 g(CHUNKS, ROWGROUPS);
  passA_kernel<<<g, 256, 0, stream>>>(feats, queue, qptr, qn2, qlab, tn2, rowmax, topk);
  passB_kernel<<<1, 256, 0, stream>>>(topk, rowmax, pseudo, Mrow);
  passC_kernel<<<g, 256, 0, stream>>>(feats, queue, qptr, qn2, qlab, tn2, pseudo, Mrow,
                                      sumall, sumpos);
  passD_kernel<<<1, 256, 0, stream>>>(sumall, sumpos, (float*)d_out);
}
